// SelfAttention_91070486544896
// MI455X (gfx1250) — matrix operand order repaired and checked by reading, not compiled, not run on hardware
//
#include <hip/hip_runtime.h>
#include <hip/hip_bf16.h>

typedef __attribute__((ext_vector_type(16))) _Float16 v16h;
typedef __attribute__((ext_vector_type(8)))  float    v8f;

#define NTOK  4096   // H*W
#define CFULL 256    // channels
#define CQ    32     // query/key channels
#define NB    4      // batch

// ---------------------------------------------------------------------------
// Kernel 1: Q/K projections (small: 0.5 GFLOP) on VALU.
// Output token-major f16: q16[b][n][32], k16[b][n][32]  (WMMA-A/B-friendly)
// ---------------------------------------------------------------------------
__global__ __launch_bounds__(256) void qk_proj_kernel(
    const float* __restrict__ x,  const float* __restrict__ wq,
    const float* __restrict__ bq, const float* __restrict__ wk,
    const float* __restrict__ bk,
    _Float16* __restrict__ q16, _Float16* __restrict__ k16)
{
  __shared__ __align__(16) float wqs[CQ * CFULL];  // transposed [c][o]
  __shared__ __align__(16) float wks[CQ * CFULL];
  const int tid = threadIdx.x;
  const int b   = blockIdx.x / (NTOK / 256);
  const int n0  = (blockIdx.x % (NTOK / 256)) * 256;

  for (int i = tid; i < CQ * CFULL; i += 256) {
    int o = i / CFULL, c = i % CFULL;
    wqs[c * CQ + o] = wq[o * CFULL + c];
    wks[c * CQ + o] = wk[o * CFULL + c];
  }
  __syncthreads();

  const int n = n0 + tid;
  float aq[CQ], ak[CQ];
#pragma unroll
  for (int o = 0; o < CQ; ++o) { aq[o] = 0.f; ak[o] = 0.f; }

  const float* xp = x + (size_t)b * CFULL * NTOK + n;
  for (int c = 0; c < CFULL; ++c) {
    float xv = xp[(size_t)c * NTOK];            // coalesced across threads
#pragma unroll
    for (int o = 0; o < CQ; ++o) {
      aq[o] = fmaf(wqs[c * CQ + o], xv, aq[o]); // LDS broadcast reads
      ak[o] = fmaf(wks[c * CQ + o], xv, ak[o]);
    }
  }
  _Float16* qr = q16 + ((size_t)b * NTOK + n) * CQ;
  _Float16* kr = k16 + ((size_t)b * NTOK + n) * CQ;
#pragma unroll
  for (int o = 0; o < CQ; ++o) {
    qr[o] = (_Float16)(aq[o] + bq[o]);
    kr[o] = (_Float16)(ak[o] + bk[o]);
  }
}

// ---------------------------------------------------------------------------
// Kernel 2: V projection as WMMA GEMM  V[b][co][n] = sum_c wv[co][c]*x[b][c][n]
// A = Wv tile (16co x 32c), B = X tile (32c x 16n) staged transposed in LDS.
// Output channel-major f16 v16[b][co][N] -> P*V tiles stage transposed free.
// ---------------------------------------------------------------------------
__global__ __launch_bounds__(256) void v_proj_kernel(
    const float* __restrict__ x, const float* __restrict__ wv,
    const float* __restrict__ bv, _Float16* __restrict__ v16)
{
  __shared__ __align__(16) _Float16 xs[16 * 40];  // [n_local][c_local], padded
  const int tid  = threadIdx.x;
  const int wid  = tid >> 5;
  const int lane = tid & 31;
  const int row  = lane & 15;
  const int hi   = lane >> 4;
  const int b    = blockIdx.x / (NTOK / 16);
  const int n0   = (blockIdx.x % (NTOK / 16)) * 16;
  const int co0  = wid * 32;                      // two 16-row co tiles/wave

  v8f acc0 = {}, acc1 = {};
  for (int kk = 0; kk < CFULL; kk += 32) {
    __syncthreads();
    for (int e = tid; e < 512; e += 256) {        // transpose-stage X tile
      int nl = e & 15, cl = e >> 4;
      xs[nl * 40 + cl] =
          (_Float16)x[((size_t)b * CFULL + kk + cl) * NTOK + n0 + nl];
    }
    __syncthreads();

    union { v16h v; uint4 u[2]; } bf;             // B frag: lane=col n, K=c
    {
      const _Float16* p = &xs[row * 40 + hi * 8];
      bf.u[0] = *(const uint4*)p;
      bf.u[1] = *(const uint4*)(p + 16);
    }
    union { v16h v; _Float16 h[16]; } a0, a1;     // A frags from wv (f32->f16)
    {
      const int c0 = kk + hi * 8;                 // K = {c0..c0+7, c0+16..+23}
      const float* w0 = wv + (size_t)(co0 + row) * CFULL + c0;
      const float* w1 = wv + (size_t)(co0 + 16 + row) * CFULL + c0;
#pragma unroll
      for (int i = 0; i < 8; ++i) {
        a0.h[i] = (_Float16)w0[i];  a0.h[i + 8] = (_Float16)w0[i + 16];
        a1.h[i] = (_Float16)w1[i];  a1.h[i + 8] = (_Float16)w1[i + 16];
      }
    }
    acc0 = __builtin_amdgcn_wmma_f32_16x16x32_f16(false, a0.v, false, bf.v,
                                                  (short)0, acc0, false, false);
    acc1 = __builtin_amdgcn_wmma_f32_16x16x32_f16(false, a1.v, false, bf.v,
                                                  (short)0, acc1, false, false);
  }
#pragma unroll
  for (int r = 0; r < 8; ++r) {                   // D: M=co_local, N=token
    int m  = r + 8 * hi;
    int co = co0 + m;
    v16[((size_t)b * CFULL + co) * NTOK + n0 + row] =
        (_Float16)(acc0[r] + bv[co]);
    v16[((size_t)b * CFULL + co + 16) * NTOK + n0 + row] =
        (_Float16)(acc1[r] + bv[co + 16]);
  }
}

// ---------------------------------------------------------------------------
// Kernel 3: flash attention. 8 waves/block; wave owns 16 query rows.
// Double-buffered K/V tiles filled by GLOBAL_LOAD_ASYNC_TO_LDS (ASYNCcnt),
// prefetch of tile j+1 overlaps the 18 WMMAs of tile j.
// acc = 16 chunks of v8f = full 16x256 f32 output tile in VGPRs.
// ---------------------------------------------------------------------------
__device__ __forceinline__ void async_copy_b64(uint32_t lds_off,
                                               const void* gptr) {
  asm volatile("global_load_async_to_lds_b64 %0, %1, off"
               :
               : "v"(lds_off), "v"((uint64_t)(uintptr_t)gptr)
               : "memory");
}
__device__ __forceinline__ void wait_async0() {
  asm volatile("s_wait_asynccnt 0x0" ::: "memory");
}

__global__ __launch_bounds__(256) void attn_kernel(
    const float* __restrict__ x,  const _Float16* __restrict__ q16,
    const _Float16* __restrict__ k16, const _Float16* __restrict__ v16,
    const float* __restrict__ gamma, float* __restrict__ out)
{
  __shared__ __align__(16) _Float16 ks[2][32 * 40];     // K tile [token][cq]
  __shared__ __align__(16) _Float16 vs[2][CFULL * 40];  // V tile [c][token]
  __shared__ __align__(16) _Float16 ps[8 * 16 * 40];    // per-wave P scratch

  const int tid  = threadIdx.x;
  const int wid  = tid >> 5;
  const int lane = tid & 31;
  const int row  = lane & 15;
  const int hi   = lane >> 4;
  const int b    = blockIdx.x / (NTOK / 128);
  const int n0   = (blockIdx.x % (NTOK / 128)) * 128 + wid * 16;

  // per-thread slots for cooperative tile DMA (8B per op, 8B aligned)
  const int r8 = tid >> 3;                              // 0..31
  const int o4 = (tid & 7) * 4;                         // 0..28 (x4 f16 = 8B)
  const _Float16* kgbase = k16 + ((size_t)b * NTOK + r8) * CQ + o4;
  const _Float16* vgbase = v16 + (size_t)b * CFULL * NTOK + o4;

  union { v16h v; uint4 u[2]; } qa;                     // Q A-frag, loaded once
  {
    const _Float16* qr = q16 + ((size_t)b * NTOK + n0 + row) * CQ + hi * 8;
    qa.u[0] = *(const uint4*)qr;
    qa.u[1] = *(const uint4*)(qr + 16);
  }

  v8f acc[16];
#pragma unroll
  for (int cc = 0; cc < 16; ++cc) acc[cc] = (v8f){};
  float rmax[8], rsum[8];
#pragma unroll
  for (int r = 0; r < 8; ++r) { rmax[r] = -3.0e38f; rsum[r] = 0.f; }

  _Float16* pw = &ps[wid * 16 * 40];

  // ---- issue async DMA of K/V tile j0 into buffer buf
  auto issue_tile = [&](int buf, int j0) {
    async_copy_b64((uint32_t)(uintptr_t)&ks[buf][r8 * 40 + o4],
                   kgbase + (size_t)j0 * CQ);
#pragma unroll
    for (int it = 0; it < 8; ++it) {
      int c = it * 32 + r8;
      async_copy_b64((uint32_t)(uintptr_t)&vs[buf][c * 40 + o4],
                     vgbase + (size_t)c * NTOK + j0);
    }
  };

  int cur = 0;
  issue_tile(cur, 0);
  wait_async0();
  __syncthreads();

  for (int j0 = 0; j0 < NTOK; j0 += 32) {
    if (j0 + 32 < NTOK) issue_tile(cur ^ 1, j0 + 32);   // prefetch next tile

    const _Float16* ksb = ks[cur];
    const _Float16* vsb = vs[cur];

    // ---- scores: S = Q (16x32) . K^T (32x16) for two key half-tiles
    union { v16h v; uint4 u[2]; } kb0, kb1;
    {
      const _Float16* p0 = &ksb[row * 40 + hi * 8];
      kb0.u[0] = *(const uint4*)p0;  kb0.u[1] = *(const uint4*)(p0 + 16);
      const _Float16* p1 = &ksb[(16 + row) * 40 + hi * 8];
      kb1.u[0] = *(const uint4*)p1;  kb1.u[1] = *(const uint4*)(p1 + 16);
    }
    v8f s0 = {}, s1 = {};
    s0 = __builtin_amdgcn_wmma_f32_16x16x32_f16(false, qa.v, false, kb0.v,
                                                (short)0, s0, false, false);
    s1 = __builtin_amdgcn_wmma_f32_16x16x32_f16(false, qa.v, false, kb1.v,
                                                (short)0, s1, false, false);

    // ---- online softmax (row m = r + 8*hi; 16 lanes hold one row's 16 cols)
    float p0v[8], p1v[8], sc[8];
#pragma unroll
    for (int r = 0; r < 8; ++r) {
      float mx = fmaxf(s0[r], s1[r]);
      mx = fmaxf(mx, __shfl_xor(mx, 1, 32));
      mx = fmaxf(mx, __shfl_xor(mx, 2, 32));
      mx = fmaxf(mx, __shfl_xor(mx, 4, 32));
      mx = fmaxf(mx, __shfl_xor(mx, 8, 32));
      float nm    = fmaxf(rmax[r], mx);
      float scale = __expf(rmax[r] - nm);
      rmax[r] = nm;
      float e0 = __expf(s0[r] - nm);
      float e1 = __expf(s1[r] - nm);
      float ss = e0 + e1;
      ss += __shfl_xor(ss, 1, 32);
      ss += __shfl_xor(ss, 2, 32);
      ss += __shfl_xor(ss, 4, 32);
      ss += __shfl_xor(ss, 8, 32);
      rsum[r] = rsum[r] * scale + ss;
      sc[r] = scale;  p0v[r] = e0;  p1v[r] = e1;
    }
#pragma unroll
    for (int cc = 0; cc < 16; ++cc)
#pragma unroll
      for (int r = 0; r < 8; ++r) acc[cc][r] *= sc[r];

    // ---- re-layout P through per-wave LDS into A-fragment form
#pragma unroll
    for (int r = 0; r < 8; ++r) {
      int m = r + 8 * hi;
      pw[m * 40 + row]      = (_Float16)p0v[r];
      pw[m * 40 + 16 + row] = (_Float16)p1v[r];
    }
    union { v16h v; uint4 u[2]; } pa;
    {
      const _Float16* pr = pw + row * 40 + hi * 8;
      pa.u[0] = *(const uint4*)pr;
      pa.u[1] = *(const uint4*)(pr + 16);
    }

    // ---- O += P (16x32) . V (32x16) for all 16 channel chunks
#pragma unroll
    for (int cc = 0; cc < 16; ++cc) {
      union { v16h v; uint4 u[2]; } vb;
      const _Float16* vp = &vsb[(cc * 16 + row) * 40 + hi * 8];
      vb.u[0] = *(const uint4*)vp;
      vb.u[1] = *(const uint4*)(vp + 16);
      acc[cc] = __builtin_amdgcn_wmma_f32_16x16x32_f16(false, pa.v, false, vb.v,
                                                       (short)0, acc[cc],
                                                       false, false);
    }

    wait_async0();        // this wave's prefetch DMA landed
    __syncthreads();      // everyone done reading cur + done prefetching
    cur ^= 1;
  }

  // ---- epilogue: out = gamma * (O / rowsum) + x   (float4 stores)
  const float g = gamma[0];
  float inv[8];
#pragma unroll
  for (int r = 0; r < 8; ++r) inv[r] = 1.f / rsum[r];
  const int m0 = 8 * hi;
#pragma unroll
  for (int cc = 0; cc < 16; ++cc) {
    const int c = cc * 16 + row;
    const size_t idx = ((size_t)b * CFULL + c) * NTOK + n0 + m0;
    float4 x0 = *(const float4*)(x + idx);
    float4 x1 = *(const float4*)(x + idx + 4);
    float4 o0, o1;
    o0.x = g * acc[cc][0] * inv[0] + x0.x;
    o0.y = g * acc[cc][1] * inv[1] + x0.y;
    o0.z = g * acc[cc][2] * inv[2] + x0.z;
    o0.w = g * acc[cc][3] * inv[3] + x0.w;
    o1.x = g * acc[cc][4] * inv[4] + x1.x;
    o1.y = g * acc[cc][5] * inv[5] + x1.y;
    o1.z = g * acc[cc][6] * inv[6] + x1.z;
    o1.w = g * acc[cc][7] * inv[7] + x1.w;
    *(float4*)(out + idx)     = o0;
    *(float4*)(out + idx + 4) = o1;
  }
}

// ---------------------------------------------------------------------------
extern "C" void kernel_launch(void* const* d_in, const int* in_sizes, int n_in,
                              void* d_out, int out_size, void* d_ws,
                              size_t ws_size, hipStream_t stream) {
  (void)in_sizes; (void)n_in; (void)out_size; (void)ws_size;
  const float* x  = (const float*)d_in[0];
  const float* wq = (const float*)d_in[1];
  const float* bq = (const float*)d_in[2];
  const float* wk = (const float*)d_in[3];
  const float* bk = (const float*)d_in[4];
  const float* wv = (const float*)d_in[5];
  const float* bv = (const float*)d_in[6];
  const float* gm = (const float*)d_in[7];
  float* out = (float*)d_out;

  char* ws = (char*)d_ws;                         // needs 10 MB scratch
  const size_t qbytes = (size_t)NB * NTOK * CQ * 2;
  _Float16* q16 = (_Float16*)(ws);
  _Float16* k16 = (_Float16*)(ws + qbytes);
  _Float16* v16 = (_Float16*)(ws + 2 * qbytes);

  hipLaunchKernelGGL(qk_proj_kernel, dim3(NB * (NTOK / 256)), dim3(256), 0,
                     stream, x, wq, bq, wk, bk, q16, k16);
  hipLaunchKernelGGL(v_proj_kernel, dim3(NB * (NTOK / 16)), dim3(256), 0,
                     stream, x, wv, bv, v16);
  hipLaunchKernelGGL(attn_kernel, dim3(NB * (NTOK / 128)), dim3(256), 0,
                     stream, x, q16, k16, v16, gm, out);
}
